// CrossAttention_20873541058969
// MI455X (gfx1250) — matrix operand order repaired and checked by reading, not compiled, not run on hardware
//
#include <hip/hip_runtime.h>

// ---------------------------------------------------------------------------
// CrossAttention block for MI455X (gfx1250, wave32, WMMA).
// All matmuls use v_wmma_f32_16x16x32_bf16 (bf16 in, f32 accumulate).
// Q-projection is fused into the attention kernel: h is L2-resident (64MB
// bf16 vs 192MB L2), so recomputing per-head Q tiles from L2 is cheaper than
// a 128MB HBM round-trip through a Q buffer.
// ---------------------------------------------------------------------------

typedef __bf16 bf16_t;
typedef __attribute__((ext_vector_type(8)))  __bf16 v8bf;
typedef __attribute__((ext_vector_type(16))) __bf16 v16bf;
typedef __attribute__((ext_vector_type(8)))  float  v8f;

#define BATCH   16
#define NQ      4096      // H*W
#define CH      512
#define DCTX    768
#define NCTX    77
#define NCTX_K  80        // scores N padded to 5*16
#define NCTX_V  96        // P@V K-dim padded to 3*32
#define HEADS   8
#define HD      64
#define NGRP    32
#define CPG     16

// ---- WMMA wrappers --------------------------------------------------------

__device__ __forceinline__ v8f wmma_bf16(v16bf a, v16bf b, v8f c) {
  // D = A(16x32) * B(32x16) + C, f32 accumulate
  return __builtin_amdgcn_wmma_f32_16x16x32_bf16(
      /*neg_a=*/false, a, /*neg_b=*/false, b,
      /*c_mod=*/(short)0, c, /*reuse_a=*/false, /*reuse_b=*/false);
}

// A fragment (16x32 bf16, ISA 7.12.2): lane = hv*16 + (m%16).
//   elems 0..7  = K[kk + 8*hv .. +7],  elems 8..15 = K[kk + 16 + 8*hv .. +7]
// rowptr must already point at this lane's row.
__device__ __forceinline__ v16bf load_a_frag(const bf16_t* __restrict__ rowptr,
                                             int kk, int hv) {
  v8bf lo = *(const v8bf*)(rowptr + kk + 8 * hv);
  v8bf hi = *(const v8bf*)(rowptr + kk + 16 + 8 * hv);
  v16bf r;
#pragma unroll
  for (int i = 0; i < 8; ++i) { r[i] = lo[i]; r[i + 8] = hi[i]; }
  return r;
}

// B fragment (32x16 bf16): lane holds column n = lane%16,
//   elems 0..7 = K[kk + 8*hv .. +7], elems 8..15 = K[kk + 16 + 8*hv .. +7]. Requires BT[n][k] layout.
__device__ __forceinline__ v16bf load_b_frag(const bf16_t* __restrict__ rowptr,
                                             int kk, int hv) {
  v8bf lo = *(const v8bf*)(rowptr + kk + 8 * hv);
  v8bf hi = *(const v8bf*)(rowptr + kk + 16 + 8 * hv);
  v16bf r;
#pragma unroll
  for (int i = 0; i < 8; ++i) { r[i] = lo[i]; r[i + 8] = hi[i]; }
  return r;
}

// ---- 1) GroupNorm statistics ---------------------------------------------
// grid = BATCH*NGRP blocks, 256 threads. Reduce 4096*16 elements per group.
__global__ void gn_stats_kernel(const float* __restrict__ x,
                                float* __restrict__ meanv,
                                float* __restrict__ rstdv) {
  const int b = blockIdx.x >> 5;
  const int g = blockIdx.x & 31;
  const float* xp = x + (size_t)b * NQ * CH + g * CPG;

  float s = 0.f, s2 = 0.f;
  for (int idx = threadIdx.x; idx < NQ * (CPG / 4); idx += 256) {
    const int pos = idx >> 2, q = idx & 3;
    const float4 v = *(const float4*)(xp + (size_t)pos * CH + q * 4);
    s  += v.x + v.y + v.z + v.w;
    s2 += v.x * v.x + v.y * v.y + v.z * v.z + v.w * v.w;
  }
  __shared__ float sh[256], sh2[256];
  sh[threadIdx.x] = s; sh2[threadIdx.x] = s2;
  __syncthreads();
#pragma unroll
  for (int st = 128; st > 0; st >>= 1) {
    if (threadIdx.x < (unsigned)st) {
      sh[threadIdx.x]  += sh[threadIdx.x + st];
      sh2[threadIdx.x] += sh2[threadIdx.x + st];
    }
    __syncthreads();
  }
  if (threadIdx.x == 0) {
    const float inv = 1.0f / (float)(NQ * CPG);
    const float m = sh[0] * inv;
    const float var = sh2[0] * inv - m * m;
    meanv[blockIdx.x] = m;
    rstdv[blockIdx.x] = rsqrtf(var + 1e-5f);
  }
}

// ---- 2) Normalize + scale/bias + cast to bf16 -----------------------------
__global__ void gn_apply_kernel(const float* __restrict__ x,
                                const float* __restrict__ meanv,
                                const float* __restrict__ rstdv,
                                const float* __restrict__ scale,
                                const float* __restrict__ bias,
                                bf16_t* __restrict__ h) {
  const size_t i = ((size_t)blockIdx.x * 256 + threadIdx.x) * 8;
  const int c = (int)(i & (CH - 1));
  const int b = (int)(i >> 21);            // / (4096*512)
  const int g = c >> 4;
  const float m = meanv[b * NGRP + g];
  const float r = rstdv[b * NGRP + g];
  const float4 v0 = *(const float4*)(x + i);
  const float4 v1 = *(const float4*)(x + i + 4);
  const float4 s0 = *(const float4*)(scale + c);
  const float4 s1 = *(const float4*)(scale + c + 4);
  const float4 b0 = *(const float4*)(bias + c);
  const float4 b1 = *(const float4*)(bias + c + 4);
  v8bf o;
  o[0] = (bf16_t)((v0.x - m) * r * s0.x + b0.x);
  o[1] = (bf16_t)((v0.y - m) * r * s0.y + b0.y);
  o[2] = (bf16_t)((v0.z - m) * r * s0.z + b0.z);
  o[3] = (bf16_t)((v0.w - m) * r * s0.w + b0.w);
  o[4] = (bf16_t)((v1.x - m) * r * s1.x + b1.x);
  o[5] = (bf16_t)((v1.y - m) * r * s1.y + b1.y);
  o[6] = (bf16_t)((v1.z - m) * r * s1.z + b1.z);
  o[7] = (bf16_t)((v1.w - m) * r * s1.w + b1.w);
  *(v8bf*)(h + i) = o;
}

// ---- 3) Weight transpose + cast: out[n*K + k] = (bf16) in[k*N + n] --------
__global__ void transpose_cast_kernel(const float* __restrict__ in,
                                      bf16_t* __restrict__ out,
                                      int K, int Nd) {
  __shared__ float tile[32][33];
  const int k0 = blockIdx.x * 32, n0 = blockIdx.y * 32;
  const int tx = threadIdx.x & 31, ty = threadIdx.x >> 5;   // 256 thr: ty 0..7
#pragma unroll
  for (int r = ty; r < 32; r += 8)
    tile[r][tx] = in[(size_t)(k0 + r) * Nd + n0 + tx];
  __syncthreads();
#pragma unroll
  for (int r = ty; r < 32; r += 8)
    out[(size_t)(n0 + r) * K + k0 + tx] = (bf16_t)tile[tx][r];
}

// ---- 4) fp32 -> bf16 elementwise (context) --------------------------------
__global__ void cvt_bf16_kernel(const float* __restrict__ in,
                                bf16_t* __restrict__ out) {
  const size_t i = ((size_t)blockIdx.x * 256 + threadIdx.x) * 8;
  const float4 a = *(const float4*)(in + i);
  const float4 b = *(const float4*)(in + i + 4);
  v8bf o;
  o[0] = (bf16_t)a.x; o[1] = (bf16_t)a.y; o[2] = (bf16_t)a.z; o[3] = (bf16_t)a.w;
  o[4] = (bf16_t)b.x; o[5] = (bf16_t)b.y; o[6] = (bf16_t)b.z; o[7] = (bf16_t)b.w;
  *(v8bf*)(out + i) = o;
}

// ---- 5) K / V projection: ctx[b](77x768) @ WT(512x768)^T + bias -----------
// grid = (CH/64, BATCH), 128 threads = 4 waves; wave owns one 16-col N-tile,
// all 5 padded M-tiles (rows clamped to 76; pad rows masked later / x0).
// STORE_VT=0: K -> Kp[b][j][c] row-major (80x512)
// STORE_VT=1: V -> Vt[b][c][j] transposed (512x96), cols 80..95 zeroed.
template <bool STORE_VT>
__global__ void gemm_kv_kernel(const bf16_t* __restrict__ ctx,
                               const bf16_t* __restrict__ BT,
                               const float* __restrict__ bias,
                               bf16_t* __restrict__ out) {
  const int b    = blockIdx.y;
  const int wave = threadIdx.x >> 5;
  const int lane = threadIdx.x & 31;
  const int hv   = lane >> 4, ln = lane & 15;
  const int nb   = blockIdx.x * 64 + wave * 16;

  const bf16_t* Ab   = ctx + (size_t)b * NCTX * DCTX;
  const bf16_t* Brow = BT + (size_t)(nb + ln) * DCTX;

  v8f acc[5] = {};
  for (int kk = 0; kk < DCTX; kk += 32) {
    const v16bf bf = load_b_frag(Brow, kk, hv);
#pragma unroll
    for (int mt = 0; mt < 5; ++mt) {
      int row = mt * 16 + ln;
      row = row > (NCTX - 1) ? (NCTX - 1) : row;   // clamp pad rows (masked later)
      const v16bf af = load_a_frag(Ab + (size_t)row * DCTX, kk, hv);
      acc[mt] = wmma_bf16(af, bf, acc[mt]);
    }
  }
  const float bv = bias[nb + ln];
  if (STORE_VT) {
    bf16_t* Vb = out + ((size_t)b * CH + nb + ln) * NCTX_V;
#pragma unroll
    for (int mt = 0; mt < 5; ++mt) {
      v8bf pk;
#pragma unroll
      for (int r = 0; r < 8; ++r) pk[r] = (bf16_t)(acc[mt][r] + bv);
      *(v8bf*)(Vb + mt * 16 + 8 * hv) = pk;        // 8 consecutive j per lane
    }
    if (hv == 0) {                                  // zero pad cols 80..95
      v8bf z;
#pragma unroll
      for (int r = 0; r < 8; ++r) z[r] = (bf16_t)0.0f;
      *(v8bf*)(Vb + 80) = z;
      *(v8bf*)(Vb + 88) = z;
    }
  } else {
    bf16_t* Kb = out + (size_t)b * NCTX_K * CH;
#pragma unroll
    for (int mt = 0; mt < 5; ++mt)
#pragma unroll
      for (int r = 0; r < 8; ++r) {
        const int j = mt * 16 + r + 8 * hv;
        Kb[(size_t)j * CH + nb + ln] = (bf16_t)(acc[mt][r] + bv);
      }
  }
}

// ---- 6) Output GEMM: A(Mx512) @ BT(512x512)^T + bias + residual -> f32 ----
// grid = (CH/64, M/128), 256 threads = 8 waves; wave (wm,wn) owns 32x32.
__global__ void gemm_out_kernel(const bf16_t* __restrict__ A,
                                const bf16_t* __restrict__ BT,
                                const float* __restrict__ bias,
                                const float* __restrict__ res,
                                float* __restrict__ out, int Kd) {
  const int wave = threadIdx.x >> 5;
  const int lane = threadIdx.x & 31;
  const int hv = lane >> 4, ln = lane & 15;
  const int wm = wave >> 1, wn = wave & 1;
  const int mb = blockIdx.y * 128 + wm * 32;
  const int nb = blockIdx.x * 64 + wn * 32;

  const bf16_t* Ar0 = A + (size_t)(mb + ln) * Kd;
  const bf16_t* Ar1 = Ar0 + (size_t)16 * Kd;
  const bf16_t* Br0 = BT + (size_t)(nb + ln) * Kd;
  const bf16_t* Br1 = Br0 + (size_t)16 * Kd;

  v8f c00 = {}, c01 = {}, c10 = {}, c11 = {};
  for (int kk = 0; kk < Kd; kk += 32) {
    const v16bf a0 = load_a_frag(Ar0, kk, hv);
    const v16bf a1 = load_a_frag(Ar1, kk, hv);
    const v16bf b0 = load_b_frag(Br0, kk, hv);
    const v16bf b1 = load_b_frag(Br1, kk, hv);
    c00 = wmma_bf16(a0, b0, c00);
    c01 = wmma_bf16(a0, b1, c01);
    c10 = wmma_bf16(a1, b0, c10);
    c11 = wmma_bf16(a1, b1, c11);
  }

  const float bv0 = bias[nb + ln];
  const float bv1 = bias[nb + 16 + ln];
#pragma unroll
  for (int r = 0; r < 8; ++r) {
    const int m0 = mb + r + 8 * hv;
    const int m1 = m0 + 16;
    const size_t i00 = (size_t)m0 * CH + nb + ln;
    const size_t i01 = (size_t)m0 * CH + nb + 16 + ln;
    const size_t i10 = (size_t)m1 * CH + nb + ln;
    const size_t i11 = (size_t)m1 * CH + nb + 16 + ln;
    out[i00] = c00[r] + bv0 + res[i00];
    out[i01] = c01[r] + bv1 + res[i01];
    out[i10] = c10[r] + bv0 + res[i10];
    out[i11] = c11[r] + bv1 + res[i11];
  }
}

// ---- 7) Fused Q-proj + attention per (64-query tile, head, batch) ---------
// 128 threads = 4 waves; wave w owns query rows 16w..16w+15.
//   phase 0: Qtile(64x64) = h_tile(64x512) @ wqT[head](64x512)^T + bq -> LDS
//            (h and wqT are L2-resident; avoids the Q HBM round-trip)
//   phase 1: scores = Qtile @ K^T -> 64x80 (WMMA, A-frags from LDS)
//   phase 2: masked softmax in LDS (pads j>=77 -> 0)
//   phase 3: out = P(64x96) @ V -> 64x64 (WMMA, A-frags from LDS)
__global__ void attention_fused_kernel(const bf16_t* __restrict__ h,
                                       const bf16_t* __restrict__ wqT,
                                       const float* __restrict__ bq,
                                       const bf16_t* __restrict__ Kp,
                                       const bf16_t* __restrict__ Vt,
                                       bf16_t* __restrict__ O) {
  const int qt = blockIdx.x, hd = blockIdx.y, b = blockIdx.z;
  const int wave = threadIdx.x >> 5;
  const int lane = threadIdx.x & 31;
  const int hv = lane >> 4, ln = lane & 15;

  const bf16_t* Kb = Kp + (size_t)b * NCTX_K * CH + hd * HD;
  const bf16_t* Vb = Vt + ((size_t)b * CH + hd * HD) * NCTX_V;

  __shared__ __align__(32) bf16_t Qls[64][HD];     // 8KB
  __shared__ __align__(32) float  S[64][NCTX_K];   // 20KB
  __shared__ __align__(32) bf16_t P[64][NCTX_V];   // 12KB

  // ---- phase 0: Q tile = h @ wq (head slice), into LDS ----
  {
    const bf16_t* hrow = h + ((size_t)b * NQ + qt * 64 + 16 * wave + ln) * CH;
    const bf16_t* wq0  = wqT + (size_t)(hd * HD) * CH;   // rows n = d of head
    v8f qc[4] = {};
    for (int kk = 0; kk < CH; kk += 32) {
      const v16bf af = load_a_frag(hrow, kk, hv);
#pragma unroll
      for (int nt = 0; nt < 4; ++nt) {
        const v16bf bf = load_b_frag(wq0 + (size_t)(nt * 16 + ln) * CH, kk, hv);
        qc[nt] = wmma_bf16(af, bf, qc[nt]);
      }
    }
#pragma unroll
    for (int nt = 0; nt < 4; ++nt) {
      const float bv = bq[hd * HD + nt * 16 + ln];
#pragma unroll
      for (int r = 0; r < 8; ++r)
        Qls[16 * wave + r + 8 * hv][nt * 16 + ln] = (bf16_t)(qc[nt][r] + bv);
    }
  }
  __syncthreads();

  // ---- phase 1: scores = Q @ K^T ----
  v8f sc[5] = {};
  const bf16_t* qrow = &Qls[16 * wave + ln][0];
#pragma unroll
  for (int kk = 0; kk < HD; kk += 32) {
    const v16bf af = load_a_frag(qrow, kk, hv);      // LDS A-fragments
#pragma unroll
    for (int nt = 0; nt < 5; ++nt) {
      const v16bf bf = load_b_frag(Kb + (size_t)(nt * 16 + ln) * CH, kk, hv);
      sc[nt] = wmma_bf16(af, bf, sc[nt]);
    }
  }
  const float scale = 0.125f;  // 1/sqrt(64)
#pragma unroll
  for (int nt = 0; nt < 5; ++nt)
#pragma unroll
    for (int r = 0; r < 8; ++r)
      S[16 * wave + r + 8 * hv][nt * 16 + ln] = sc[nt][r] * scale;
  __syncthreads();

  // ---- phase 2: masked softmax (one thread per query row) ----
  if (threadIdx.x < 64) {
    const int row = threadIdx.x;
    float mx = -1e30f;
    for (int j = 0; j < NCTX; ++j) mx = fmaxf(mx, S[row][j]);
    float sum = 0.f;
    for (int j = 0; j < NCTX; ++j) {
      const float e = __expf(S[row][j] - mx);
      S[row][j] = e;
      sum += e;
    }
    const float inv = 1.0f / sum;
    for (int j = 0; j < NCTX; ++j) P[row][j] = (bf16_t)(S[row][j] * inv);
    for (int j = NCTX; j < NCTX_V; ++j) P[row][j] = (bf16_t)0.0f;  // mask pads
  }
  __syncthreads();

  // ---- phase 3: out = P @ V ----
  v8f oc[4] = {};
  const bf16_t* prow = &P[16 * wave + ln][0];
#pragma unroll
  for (int kk = 0; kk < NCTX_V; kk += 32) {
    const v16bf af = load_a_frag(prow, kk, hv);      // LDS A-fragments
#pragma unroll
    for (int nt = 0; nt < 4; ++nt) {
      const v16bf bf = load_b_frag(Vb + (size_t)(nt * 16 + ln) * NCTX_V, kk, hv);
      oc[nt] = wmma_bf16(af, bf, oc[nt]);
    }
  }
  bf16_t* Ob = O + ((size_t)b * NQ + qt * 64) * CH + hd * HD;
#pragma unroll
  for (int nt = 0; nt < 4; ++nt)
#pragma unroll
    for (int r = 0; r < 8; ++r)
      Ob[(size_t)(16 * wave + r + 8 * hv) * CH + nt * 16 + ln] = (bf16_t)oc[nt][r];
}

// ---------------------------------------------------------------------------
extern "C" void kernel_launch(void* const* d_in, const int* in_sizes, int n_in,
                              void* d_out, int out_size, void* d_ws, size_t ws_size,
                              hipStream_t stream) {
  const float* x   = (const float*)d_in[0];
  const float* ctx = (const float*)d_in[1];
  const float* gns = (const float*)d_in[2];
  const float* gnb = (const float*)d_in[3];
  const float* wq  = (const float*)d_in[4];
  const float* bq  = (const float*)d_in[5];
  const float* wk  = (const float*)d_in[6];
  const float* bk  = (const float*)d_in[7];
  const float* wv  = (const float*)d_in[8];
  const float* bv  = (const float*)d_in[9];
  const float* wo  = (const float*)d_in[10];
  const float* bo  = (const float*)d_in[11];
  float* out = (float*)d_out;

  char* ws = (char*)d_ws;
  auto alloc = [&](size_t bytes) -> char* {
    char* p = ws;
    ws += (bytes + 255) & ~(size_t)255;
    return p;
  };
  float*  meanv = (float*)alloc(BATCH * NGRP * 4);
  float*  rstdv = (float*)alloc(BATCH * NGRP * 4);
  bf16_t* wqT   = (bf16_t*)alloc((size_t)CH * CH * 2);
  bf16_t* wkT   = (bf16_t*)alloc((size_t)CH * DCTX * 2);
  bf16_t* wvT   = (bf16_t*)alloc((size_t)CH * DCTX * 2);
  bf16_t* woT   = (bf16_t*)alloc((size_t)CH * CH * 2);
  bf16_t* ctxb  = (bf16_t*)alloc((size_t)BATCH * NCTX * DCTX * 2);
  bf16_t* Kpad  = (bf16_t*)alloc((size_t)BATCH * NCTX_K * CH * 2);
  bf16_t* Vtr   = (bf16_t*)alloc((size_t)BATCH * CH * NCTX_V * 2);
  bf16_t* hb    = (bf16_t*)alloc((size_t)BATCH * NQ * CH * 2);  // normalized h
  bf16_t* attnо = (bf16_t*)alloc((size_t)BATCH * NQ * CH * 2);  // attention out
  bf16_t* attno = attnо;

  // 1-2) GroupNorm
  gn_stats_kernel<<<BATCH * NGRP, 256, 0, stream>>>(x, meanv, rstdv);
  gn_apply_kernel<<<(BATCH * (size_t)NQ * CH) / (256 * 8), 256, 0, stream>>>(
      x, meanv, rstdv, gns, gnb, hb);

  // 3-4) Weight transposes + context cast
  transpose_cast_kernel<<<dim3(CH / 32, CH / 32), 256, 0, stream>>>(wq, wqT, CH, CH);
  transpose_cast_kernel<<<dim3(DCTX / 32, CH / 32), 256, 0, stream>>>(wk, wkT, DCTX, CH);
  transpose_cast_kernel<<<dim3(DCTX / 32, CH / 32), 256, 0, stream>>>(wv, wvT, DCTX, CH);
  transpose_cast_kernel<<<dim3(CH / 32, CH / 32), 256, 0, stream>>>(wo, woT, CH, CH);
  cvt_bf16_kernel<<<((size_t)BATCH * NCTX * DCTX) / (256 * 8), 256, 0, stream>>>(ctx, ctxb);

  // 5) K and V projections
  gemm_kv_kernel<false><<<dim3(CH / 64, BATCH), 128, 0, stream>>>(ctxb, wkT, bk, Kpad);
  gemm_kv_kernel<true ><<<dim3(CH / 64, BATCH), 128, 0, stream>>>(ctxb, wvT, bv, Vtr);

  // 6) Fused Q-projection + attention
  attention_fused_kernel<<<dim3(NQ / 64, HEADS, BATCH), 128, 0, stream>>>(
      hb, wqT, bq, Kpad, Vtr, attno);

  // 7) Output projection + bias + residual -> fp32 out
  gemm_out_kernel<<<dim3(CH / 64, (BATCH * NQ) / 128), 256, 0, stream>>>(
      attno, woT, bo, x, out, CH);
}
